// SpMiddlePillarEncoder8xHA_43817256354291
// MI455X (gfx1250) — compile-verified, matrix operand order checked by reading, not run on hardware
//
#include <hip/hip_runtime.h>
#include <hip/hip_bf16.h>

#define PC_MIN_ (-75.2f)
#define BEV_    (0.8f)
#define GRID_   188
#define H5_     94

typedef __attribute__((ext_vector_type(16))) __bf16        bf16x16;
typedef __attribute__((ext_vector_type(8)))  float         f32x8;
typedef __attribute__((ext_vector_type(4)))  unsigned int  u32x4;

union Frag { bf16x16 v; u32x4 q[2]; };

static __device__ __forceinline__ unsigned short f2bf(float x) {
  unsigned int u = __float_as_uint(x);
  u += 0x7fffu + ((u >> 16) & 1u);           // round-to-nearest-even
  return (unsigned short)(u >> 16);
}
static __device__ __forceinline__ float bf2f(unsigned short h) {
  return __uint_as_float(((unsigned int)h) << 16);
}

// ---------------- utility kernels ----------------
__global__ void zero_f32(float* __restrict__ p, long long n) {
  long long stride = (long long)gridDim.x * blockDim.x;
  for (long long i = (long long)blockIdx.x * blockDim.x + threadIdx.x; i < n; i += stride)
    p[i] = 0.0f;
}

__global__ void cvt_f32_bf16(const float* __restrict__ src, unsigned short* __restrict__ dst,
                             long long n) {
  long long stride = (long long)gridDim.x * blockDim.x;
  for (long long i = (long long)blockIdx.x * blockDim.x + threadIdx.x; i < n; i += stride)
    dst[i] = f2bf(src[i]);
}

// fp32 w[3][3][Cin][Cout]  ->  bf16 wt[9][Cout][Cin]   (B^T layout, ci contiguous)
__global__ void wcvt(const float* __restrict__ src, unsigned short* __restrict__ dst,
                     int Cin, int Cout, int n) {
  int i = blockIdx.x * blockDim.x + threadIdx.x;
  if (i >= n) return;
  int ci = i % Cin;
  int t  = i / Cin;
  int co = t % Cout;
  int p  = t / Cout;
  dst[i] = f2bf(src[((long long)p * Cin + ci) * Cout + co]);
}

// ---------------- pillar scatter-max + point MLP ----------------
__global__ void pillar_pool(const float* __restrict__ xyz, const int* __restrict__ bcnt,
                            const float* __restrict__ pf,  const float* __restrict__ wm,
                            const float* __restrict__ ms,  const float* __restrict__ mb,
                            float* __restrict__ pooled, float* __restrict__ maskb, int Npts) {
  long long gid = (long long)blockIdx.x * blockDim.x + threadIdx.x;
  int p    = (int)(gid >> 5);
  int lane = (int)(gid & 31);
  if (p >= Npts) return;

  float x = xyz[3 * p + 0], y = xyz[3 * p + 1], z = xyz[3 * p + 2];
  int bid = (p >= bcnt[0]) ? 1 : 0;

  int px = (int)floorf((x - PC_MIN_) / BEV_);
  int py = (int)floorf((y - PC_MIN_) / BEV_);
  px = min(max(px, 0), GRID_ - 1);
  py = min(max(py, 0), GRID_ - 1);
  float cx = PC_MIN_ + ((float)px + 0.5f) * BEV_;
  float cy = PC_MIN_ + ((float)py + 0.5f) * BEV_;

  float feat[8];
  feat[0] = x; feat[1] = y; feat[2] = z;
  feat[3] = x - cx; feat[4] = y - cy; feat[5] = z;
  feat[6] = pf[2 * p + 0]; feat[7] = pf[2 * p + 1];

  long long seg = ((long long)(bid * GRID_ + py)) * GRID_ + px;
  float* cell = pooled + seg * 128;

#pragma unroll
  for (int j = 0; j < 4; ++j) {
    int c = lane * 4 + j;
    float h = 0.0f;
#pragma unroll
    for (int k = 0; k < 8; ++k) h += feat[k] * wm[k * 128 + c];
    h = fmaxf(h * ms[c] + mb[c], 0.0f);
    atomicMax((unsigned int*)(cell + c), __float_as_uint(h));  // valid: h >= 0
  }
  if (lane == 0) maskb[seg] = 1.0f;
}

// ---------------- WMMA implicit-GEMM 3x3 conv (LDS-staged activations) ----------------
// in   : bf16 [2, Hin, Win, CIN]       wt : bf16 [9][COUT][CIN]
// out  : y = relu((conv*scale + bias) [*mask] [+res])  -> bf16 NHWC and/or fp32 NCHW
template <int CIN, int COUT, int STRIDE, int PAD>
__global__ void __launch_bounds__(256)
conv_wmma(const unsigned short* __restrict__ in, const unsigned short* __restrict__ wt,
          const float* __restrict__ scale, const float* __restrict__ bias,
          const float* __restrict__ mask, const unsigned short* __restrict__ res,
          unsigned short* __restrict__ out_bf, float* __restrict__ out_f32,
          int Hin, int Win, int Ho, int Wo) {
  constexpr int NG      = COUT / 64;          // waves along output channels
  constexpr int XTPB    = 8 / NG;             // 16-pixel x-tiles per block
  constexpr int OUTSPAN = XTPB * 16;          // output pixels covered per block
  constexpr int XSPAN   = OUTSPAN * STRIDE + 2;
  constexpr int CQ      = CIN / 8;            // 16B quads per pixel
  constexpr int NQUADS  = 3 * XSPAN * CQ;

  __shared__ __align__(16) unsigned short lds[3 * XSPAN * CIN];

  const int b   = blockIdx.z;
  const int yo  = blockIdx.y;
  const int x0g = blockIdx.x * OUTSPAN;       // block's output-x base

  // ---- stage 3 zero-padded input rows into LDS ----
  const u32x4 z4 = { 0u, 0u, 0u, 0u };
  for (int q = threadIdx.x; q < NQUADS; q += 256) {
    const int cq  = q % CQ;
    const int pxl = q / CQ;
    const int row = pxl / XSPAN;
    const int xil = pxl % XSPAN;
    const int yin = yo * STRIDE + row - PAD;
    const int xin = x0g * STRIDE - PAD + xil;
    u32x4 v = z4;
    if (yin >= 0 && yin < Hin && xin >= 0 && xin < Win)
      v = *(const u32x4*)(in + ((long long)(b * Hin + yin) * Win + xin) * CIN + cq * 8);
    *(u32x4*)(lds + (long long)q * 8) = v;
  }
  __syncthreads();

  const int lane  = threadIdx.x & 31;
  const int wv    = threadIdx.x >> 5;
  const int ng    = wv & (NG - 1);
  const int xtl   = wv / NG;
  const int x0    = x0g + xtl * 16;
  const int nbase = ng * 64;
  const int mcol  = lane & 15;                // A row / B col / D col
  const int khalf = (lane >> 4) << 3;         // K offset 0 or 8 (fragment layout)

  f32x8 acc[4] = { {}, {}, {}, {} };

  for (int p = 0; p < 9; ++p) {
    const int ky  = p / 3, kx = p % 3;
    const int xil = (xtl * 16 + mcol) * STRIDE + kx;
    const unsigned short* arow = lds + (ky * XSPAN + xil) * CIN;
    const unsigned short* wrow = wt + (long long)p * COUT * CIN + (nbase + mcol) * CIN;
    if (p < 8) __builtin_prefetch(wt + (long long)(p + 1) * COUT * CIN, 0, 3);
#pragma unroll
    for (int cb = 0; cb < CIN; cb += 32) {
      Frag a;
      const unsigned short* ap = arow + cb + khalf;
      a.q[0] = *(const u32x4*)(ap);           // ds_load_b128
      a.q[1] = *(const u32x4*)(ap + 16);
      Frag bf[4];
#pragma unroll
      for (int tt = 0; tt < 4; ++tt) {
        const unsigned short* w2 = wrow + tt * 16 * CIN + cb + khalf;
        bf[tt].q[0] = *(const u32x4*)(w2);    // global_load_b128 (L2-resident weights)
        bf[tt].q[1] = *(const u32x4*)(w2 + 16);
      }
#pragma unroll
      for (int tt = 0; tt < 4; ++tt)
        acc[tt] = __builtin_amdgcn_wmma_f32_16x16x32_bf16(
            false, a.v, false, bf[tt].v, (short)0, acc[tt], false, false);
    }
  }

  // -------- epilogue (always ReLU) --------
#pragma unroll
  for (int tt = 0; tt < 4; ++tt) {
    const int co = nbase + tt * 16 + mcol;
    const float sc = scale[co];
    const float bi = bias[co];
#pragma unroll
    for (int v = 0; v < 8; ++v) {
      const int mrow = ((lane >> 4) << 3) + v;          // D: M = (lane/16)*8 + vgpr
      const int xo = x0 + mrow;
      if (xo >= Wo) continue;
      float y = acc[tt][v] * sc + bi;
      const long long pix = (long long)(b * Ho + yo) * Wo + xo;
      if (mask) y *= mask[pix];
      if (res)  y += bf2f(res[pix * COUT + co]);
      y = fmaxf(y, 0.0f);
      if (out_bf)  out_bf[pix * COUT + co] = f2bf(y);
      if (out_f32) out_f32[((long long)(b * COUT + co) * Ho + yo) * Wo + xo] = y;
    }
  }
}

// ---------------- host launch ----------------
template <int CIN, int COUT, int STRIDE, int PAD>
static void launch_conv(const unsigned short* in, const unsigned short* wt,
                        const float* sc, const float* bi, const float* mask,
                        const unsigned short* res, unsigned short* obf, float* of32,
                        int Hin, int Win, int Ho, int Wo, hipStream_t stream) {
  constexpr int OUTSPAN = (8 / (COUT / 64)) * 16;
  dim3 g((Wo + OUTSPAN - 1) / OUTSPAN, Ho, 2);
  conv_wmma<CIN, COUT, STRIDE, PAD><<<g, 256, 0, stream>>>(
      in, wt, sc, bi, mask, res, obf, of32, Hin, Win, Ho, Wo);
}

extern "C" void kernel_launch(void* const* d_in, const int* in_sizes, int n_in,
                              void* d_out, int out_size, void* d_ws, size_t ws_size,
                              hipStream_t stream) {
  const float* xyz  = (const float*)d_in[0];
  const int*   bcnt = (const int*)d_in[1];
  const float* pf   = (const float*)d_in[2];
  const float* wm   = (const float*)d_in[3];
  const float* ms   = (const float*)d_in[4];
  const float* mb   = (const float*)d_in[5];
  const float *w4[5], *s4[5], *b4[5];
  for (int i = 0; i < 5; ++i) {
    w4[i] = (const float*)d_in[6 + 3 * i];
    s4[i] = (const float*)d_in[7 + 3 * i];
    b4[i] = (const float*)d_in[8 + 3 * i];
  }
  const float* w5d = (const float*)d_in[21];
  const float* s5d = (const float*)d_in[22];
  const float* b5d = (const float*)d_in[23];
  const float *w5[4], *s5[4], *b5[4];
  for (int i = 0; i < 4; ++i) {
    w5[i] = (const float*)d_in[24 + 3 * i];
    s5[i] = (const float*)d_in[25 + 3 * i];
    b5[i] = (const float*)d_in[26 + 3 * i];
  }
  const int Npts = in_sizes[0] / 3;

  // ---- workspace layout ----
  char* ws = (char*)d_ws;
  const long long POOLED_F     = 2LL * GRID_ * GRID_ * 128;           // floats
  const long long MASK_F       = 2LL * GRID_ * GRID_;
  const long long POOLED_BYTES = POOLED_F * 4;
  const long long MASK_OFF     = POOLED_BYTES;
  const long long A_OFF        = MASK_OFF + MASK_F * 4;
  const long long ASLOT        = 2LL * GRID_ * GRID_ * 128 * 2;       // bf16 bytes
  const long long W_OFF        = A_OFF + 3 * ASLOT;
  const long long BSLOT        = 2LL * H5_ * H5_ * 256 * 2;           // bf16 bytes
  const long long W4SZ  = 9LL * 128 * 128 * 2;
  const long long W5DSZ = 9LL * 128 * 256 * 2;
  const long long W5SZ  = 9LL * 256 * 256 * 2;

  float* pooled = (float*)ws;
  float* maskb  = (float*)(ws + MASK_OFF);
  unsigned short* A[3]  = { (unsigned short*)(ws + A_OFF),
                            (unsigned short*)(ws + A_OFF + ASLOT),
                            (unsigned short*)(ws + A_OFF + 2 * ASLOT) };
  unsigned short* Bb[3] = { (unsigned short*)ws,                      // reuses dead pooled region
                            (unsigned short*)(ws + BSLOT),
                            (unsigned short*)(ws + 2 * BSLOT) };
  unsigned short* W4[5];
  for (int i = 0; i < 5; ++i) W4[i] = (unsigned short*)(ws + W_OFF + i * W4SZ);
  unsigned short* W5D = (unsigned short*)(ws + W_OFF + 5 * W4SZ);
  unsigned short* W5[4];
  for (int i = 0; i < 4; ++i)
    W5[i] = (unsigned short*)(ws + W_OFF + 5 * W4SZ + W5DSZ + i * W5SZ);

  // 1. zero pooled grid + mask
  zero_f32<<<4096, 256, 0, stream>>>(pooled, POOLED_F + MASK_F);

  // 2. convert conv weights to bf16 [tap][co][ci]
  for (int i = 0; i < 5; ++i) {
    int n = 9 * 128 * 128;
    wcvt<<<(n + 255) / 256, 256, 0, stream>>>(w4[i], W4[i], 128, 128, n);
  }
  { int n = 9 * 128 * 256; wcvt<<<(n + 255) / 256, 256, 0, stream>>>(w5d, W5D, 128, 256, n); }
  for (int i = 0; i < 4; ++i) {
    int n = 9 * 256 * 256;
    wcvt<<<(n + 255) / 256, 256, 0, stream>>>(w5[i], W5[i], 256, 256, n);
  }

  // 3. pillar pooling (point MLP + scatter-max + mask)
  long long pth = (long long)Npts * 32;
  pillar_pool<<<(int)((pth + 255) / 256), 256, 0, stream>>>(xyz, bcnt, pf, wm, ms, mb,
                                                            pooled, maskb, Npts);

  // 4. pooled fp32 -> bf16 activation A0
  cvt_f32_bf16<<<4096, 256, 0, stream>>>(pooled, A[0], POOLED_F);

  float* out4 = (float*)d_out;                            // [2,128,188,188]
  float* out5 = out4 + 2LL * 128 * GRID_ * GRID_;         // [2,256,94,94]

  // ---- conv4 (submanifold, masked) ----
  launch_conv<128,128,1,1>(A[0], W4[0], s4[0], b4[0], maskb, nullptr, A[1], nullptr,
                           GRID_, GRID_, GRID_, GRID_, stream);   // x1 = relu(subm(A0))
  launch_conv<128,128,1,1>(A[1], W4[1], s4[1], b4[1], maskb, nullptr, A[0], nullptr,
                           GRID_, GRID_, GRID_, GRID_, stream);   // t  = relu(subm(x1))
  launch_conv<128,128,1,1>(A[0], W4[2], s4[2], b4[2], maskb, A[1], A[2], nullptr,
                           GRID_, GRID_, GRID_, GRID_, stream);   // x2 = relu(subm(t)+x1)
  launch_conv<128,128,1,1>(A[2], W4[3], s4[3], b4[3], maskb, nullptr, A[0], nullptr,
                           GRID_, GRID_, GRID_, GRID_, stream);   // t  = relu(subm(x2))
  launch_conv<128,128,1,1>(A[0], W4[4], s4[4], b4[4], maskb, A[2], A[1], out4,
                           GRID_, GRID_, GRID_, GRID_, stream);   // x_conv4 (+fp32 NCHW)

  // ---- conv5 (dense) ----
  launch_conv<128,256,2,0>(A[1], W5D, s5d, b5d, nullptr, nullptr, Bb[0], nullptr,
                           GRID_, GRID_, H5_, H5_, stream);       // stride-2 down
  launch_conv<256,256,1,1>(Bb[0], W5[0], s5[0], b5[0], nullptr, nullptr, Bb[1], nullptr,
                           H5_, H5_, H5_, H5_, stream);           // t = relu(conv(x))
  launch_conv<256,256,1,1>(Bb[1], W5[1], s5[1], b5[1], nullptr, Bb[0], Bb[2], nullptr,
                           H5_, H5_, H5_, H5_, stream);           // x = relu(conv(t)+idn)
  launch_conv<256,256,1,1>(Bb[2], W5[2], s5[2], b5[2], nullptr, nullptr, Bb[1], nullptr,
                           H5_, H5_, H5_, H5_, stream);           // t = relu(conv(x))
  launch_conv<256,256,1,1>(Bb[1], W5[3], s5[3], b5[3], nullptr, Bb[2], nullptr, out5,
                           H5_, H5_, H5_, H5_, stream);           // x_conv5 (fp32 NCHW)
}
